// RsVocoderSynthesisLstmModel_89369679495190
// MI455X (gfx1250) — matrix operand order repaired and checked by reading, not compiled, not run on hardware
//
#include <hip/hip_runtime.h>
#include <hip/hip_bf16.h>

// ---------------------------------------------------------------------------
// 3-layer LSTM (B=32, T=1024, H=512) for gfx1250.
//  Phase 1 (parallel): XP = x @ W_ih^T + b  via bf16 WMMA GEMM.
//  Phase 2 (sequential): persistent 16-WG kernel, W_hh slice resident in LDS
//  (loaded via TDM tensor_load_to_lds), per-step h tile staged via TDM,
//  one grid sync per timestep (cluster barrier + atomic fallback).
// ---------------------------------------------------------------------------

typedef __bf16 bf16_t;
typedef __attribute__((ext_vector_type(16))) __bf16 v16bf;
typedef __attribute__((ext_vector_type(8)))  __bf16 v8bf;
typedef __attribute__((ext_vector_type(8)))  float  v8f;
typedef __attribute__((ext_vector_type(4)))  unsigned int u32x4;
typedef __attribute__((ext_vector_type(4)))  int i32x4;
typedef __attribute__((ext_vector_type(8)))  int i32x8;

constexpr int NB = 32;    // batch
constexpr int NC = 4;     // channels
constexpr int NT = 1024;  // time
constexpr int NF = 128;   // features per channel
constexpr int NI = 512;   // input size (NC*NF)
constexpr int NH = 512;   // hidden
constexpr int NG = 2048;  // 4*NH gates
constexpr int NO = 80;    // output
constexpr int BT = NB * NT;  // rows, m = t*NB + b

#define CAT16(lo, hi) \
  __builtin_shufflevector(lo, hi, 0, 1, 2, 3, 4, 5, 6, 7, 8, 9, 10, 11, 12, 13, 14, 15)

#if defined(__has_builtin)
#if __has_builtin(__builtin_amdgcn_tensor_load_to_lds)
#define HAVE_TDM 1
#endif
#endif
#ifndef HAVE_TDM
#define HAVE_TDM 0
#endif

static __device__ __forceinline__ v8f wmma_bf16(v16bf a, v16bf b, v8f c) {
  return __builtin_amdgcn_wmma_f32_16x16x32_bf16(false, a, false, b, (short)0, c,
                                                 false, false);
}

static __device__ __forceinline__ float fsig(float x) {
  x = fminf(fmaxf(x, -30.f), 30.f);
  return 1.0f / (1.0f + __expf(-x));
}
static __device__ __forceinline__ float ftanh_(float x) {
  x = fminf(fmaxf(x, -15.f), 15.f);
  float e = __expf(2.0f * x);
  return (e - 1.0f) / (e + 1.0f);
}

#if HAVE_TDM
// Issue one TDM tensor load: up to 3-D tile of bf16 (data_size=2B) from global
// into LDS at byte offset lds_off. Dims/strides in elements, per ISA 08 §8.
static __device__ __forceinline__ void tdm_load_bf16(
    const bf16_t* gaddr, unsigned lds_off, unsigned tile_d0, unsigned tile_d1,
    unsigned tile_d2, unsigned tensor_d0, unsigned tensor_d1, unsigned tensor_d2,
    unsigned long long d0_stride, unsigned long long d1_stride) {
  unsigned long long ga = (unsigned long long)(const void*)gaddr;
  u32x4 g0 = {1u,  // count=1 valid descriptor, user mode, no gather
              lds_off, (unsigned)(ga & 0xffffffffu),
              (unsigned)((ga >> 32) & 0x1ffffffu) | (2u << 30)};  // type=2
  i32x8 g1;
  g1[0] = (int)(1u << 16);  // workgroup_mask=0, data_size=1 (2 bytes)
  g1[1] = (int)((tensor_d0 & 0xffffu) << 16);
  g1[2] = (int)(((tensor_d0 >> 16) & 0xffffu) | ((tensor_d1 & 0xffffu) << 16));
  g1[3] = (int)(((tensor_d1 >> 16) & 0xffffu) | ((tile_d0 & 0xffffu) << 16));
  g1[4] = (int)((tile_d1 & 0xffffu) | ((tile_d2 & 0xffffu) << 16));
  g1[5] = (int)(unsigned)(d0_stride & 0xffffffffu);
  g1[6] = (int)(((unsigned)(d0_stride >> 32) & 0xffffu) |
                ((unsigned)(d1_stride & 0xffffu) << 16));
  g1[7] = (int)(unsigned)((d1_stride >> 16) & 0xffffffffu);
  i32x4 g2 = {(int)tensor_d2, 0, 0, 0};  // tensor_dim2 (0 for 2-D)
  i32x4 g3 = {0, 0, 0, 0};
#if __clang_major__ >= 23
  i32x8 g4 = {0, 0, 0, 0, 0, 0, 0, 0};
  __builtin_amdgcn_tensor_load_to_lds(g0, g1, g2, g3, g4, 0);
#else
  __builtin_amdgcn_tensor_load_to_lds(g0, g1, g2, g3, 0);
#endif
}
#endif

// Grid sync across the 16 persistent WGs. The cluster barrier is an
// architected NOP when not launched as a cluster (ClusterID==0); the atomic
// monotonic counter guarantees the sync either way.
static __device__ __forceinline__ void grid_sync(unsigned* bar, unsigned& goal) {
  __threadfence();
  __syncthreads();
  __builtin_amdgcn_s_cluster_barrier();
  goal += 16u;
  if (threadIdx.x == 0) {
    __hip_atomic_fetch_add(bar, 1u, __ATOMIC_ACQ_REL, __HIP_MEMORY_SCOPE_AGENT);
    while (__hip_atomic_load(bar, __ATOMIC_ACQUIRE, __HIP_MEMORY_SCOPE_AGENT) < goal) {
      __builtin_amdgcn_s_sleep(8);
    }
  }
  __syncthreads();
}

// --------------------------- small helper kernels ---------------------------

__global__ void k_init(unsigned* bar) {
  if (threadIdx.x < 64) bar[threadIdx.x] = 0u;
}

// X[B,C,T,F] fp32 -> xin[(t*NB+b), c*NF+f] bf16
__global__ void k_reshape_x(const float* __restrict__ X, bf16_t* __restrict__ xin) {
  int n = blockIdx.x * blockDim.x + threadIdx.x;
  int f = n & (NF - 1);
  int t = (n >> 7) & (NT - 1);
  int c = (n >> 17) & (NC - 1);
  int b = n >> 19;
  xin[(size_t)(t * NB + b) * NI + c * NF + f] = (bf16_t)X[n];
}

__global__ void k_f32_to_bf16(const float* __restrict__ s, bf16_t* __restrict__ d,
                              int n) {
  int i = blockIdx.x * blockDim.x + threadIdx.x;
  if (i < n) d[i] = (bf16_t)s[i];
}

__global__ void k_bias(const float* __restrict__ a, const float* __restrict__ b,
                       float* __restrict__ c) {
  int g = blockIdx.x * blockDim.x + threadIdx.x;
  if (g < NG) c[g] = a[g] + b[g];
}

// ------------------------- input-projection GEMM ---------------------------
// XP[m, g] = sum_k A[m,k] * Wih[g,k] + bias[g]   (A,Wih bf16; XP f32)
// Block tile 64(M) x 128(N); 8 waves in 2x4; 4 wmma tiles per wave.
__global__ void k_xproj(const bf16_t* __restrict__ A,
                        const bf16_t* __restrict__ Wih,
                        const float* __restrict__ bias, float* __restrict__ XP) {
  __shared__ __align__(16) bf16_t As[64 * 32];
  __shared__ __align__(16) bf16_t Bs[128 * 32];
  const int m0 = blockIdx.x * 64;
  const int n0 = blockIdx.y * 128;
  const int tid = threadIdx.x;
  const int lane = tid & 31;
  const int w = tid >> 5;
  const int mw = w & 1;   // 0..1 -> 32 rows
  const int nw = w >> 1;  // 0..3 -> 32 cols
  const int l15 = lane & 15;
  const int lh = lane >> 4;  // 0/1

  v8f acc00 = {}, acc01 = {}, acc10 = {}, acc11 = {};

  const int ar = tid >> 2, as_ = tid & 3;   // A staging: 64 rows x 4 segs
  const int br = tid >> 1, bs_ = tid & 1;   // B staging: 128 rows x 2 segs

  for (int k0 = 0; k0 < NI; k0 += 32) {
    __syncthreads();
    *(v8bf*)&As[ar * 32 + as_ * 8] =
        *(const v8bf*)&A[(size_t)(m0 + ar) * NI + k0 + as_ * 8];
    const bf16_t* wsrc = &Wih[(size_t)(n0 + br) * NI + k0 + bs_ * 16];
    *(v8bf*)&Bs[br * 32 + bs_ * 16] = *(const v8bf*)(wsrc);
    *(v8bf*)&Bs[br * 32 + bs_ * 16 + 8] = *(const v8bf*)(wsrc + 8);
    if (k0 + 32 < NI) {
      __builtin_prefetch(&A[(size_t)(m0 + ar) * NI + k0 + 32 + as_ * 8], 0, 1);
      __builtin_prefetch(&Wih[(size_t)(n0 + br) * NI + k0 + 32 + bs_ * 16], 0, 1);
    }
    __syncthreads();

    const int kb = lh * 8;    // A lane K base (ISA 16-bit A layout)
    const int kb2 = lh * 8;   // B lane K base (same K order as A)
    const bf16_t* p = &As[(mw * 32 + l15) * 32 + kb];
    v16bf a0 = CAT16(*(const v8bf*)p, *(const v8bf*)(p + 16));
    const bf16_t* q = &As[(mw * 32 + 16 + l15) * 32 + kb];
    v16bf a1 = CAT16(*(const v8bf*)q, *(const v8bf*)(q + 16));
    const bf16_t* r0 = &Bs[(nw * 32 + l15) * 32 + kb2];
    v16bf b0 = CAT16(*(const v8bf*)r0, *(const v8bf*)(r0 + 16));
    const bf16_t* r1 = &Bs[(nw * 32 + 16 + l15) * 32 + kb2];
    v16bf b1 = CAT16(*(const v8bf*)r1, *(const v8bf*)(r1 + 16));

    acc00 = wmma_bf16(a0, b0, acc00);
    acc01 = wmma_bf16(a0, b1, acc01);
    acc10 = wmma_bf16(a1, b0, acc10);
    acc11 = wmma_bf16(a1, b1, acc11);
  }

#pragma unroll
  for (int r = 0; r < 8; ++r) {
    const int Ma = m0 + mw * 32 + r + 8 * lh;
    const int N0 = n0 + nw * 32 + l15;
    const int N1 = N0 + 16;
    XP[(size_t)Ma * NG + N0] = acc00[r] + bias[N0];
    XP[(size_t)Ma * NG + N1] = acc01[r] + bias[N1];
    XP[(size_t)(Ma + 16) * NG + N0] = acc10[r] + bias[N0];
    XP[(size_t)(Ma + 16) * NG + N1] = acc11[r] + bias[N1];
  }
}

// --------------------------- recurrent LSTM scan ---------------------------
// 16 persistent WGs; WG `wg` owns hidden units [wg*32, wg*32+32) across all 4
// gates. Its 128x512 W_hh slice (bf16, 128KB) lives in LDS all layer long.
__global__ void k_lstm_seq(const float* __restrict__ XP,    // [NT*NB, NG]
                           const bf16_t* __restrict__ Whh,  // [NG, NH] bf16
                           bf16_t* __restrict__ h_ping,     // [NB, NH]
                           bf16_t* __restrict__ h_pong,     // [NB, NH]
                           bf16_t* __restrict__ hs,         // [NT*NB, NH]
                           unsigned* __restrict__ bar, unsigned goal) {
  extern __shared__ char smem[];
  bf16_t* w_s = (bf16_t*)smem;                  // 128*512 bf16 = 128KB @ 0
  bf16_t* h_s = (bf16_t*)(smem + 131072);       // 32*512 bf16  = 32KB
  float* g_s = (float*)(smem + 163840);         // 32*128 f32   = 16KB
  float* c_s = (float*)(smem + 180224);         // 32*32  f32   = 4KB

  const int wg = blockIdx.x;  // 0..15
  const int j0 = wg * 32;
  const int tid = threadIdx.x;
  const int lane = tid & 31;
  const int w = tid >> 5;  // wave 0..7 -> owns 16 local gate cols
  const int l15 = lane & 15;
  const int lh = lane >> 4;

  // Preload W_hh slice: local row r -> global row (r/32)*512 + j0 + (r%32).
#if HAVE_TDM
  if (w == 0) {
    // 3-D TDM gather: X=K(512), Y=32 rows (stride 512), Z=4 gate blocks
    // (stride 512*512) -> contiguous 128KB LDS image at offset 0.
    tdm_load_bf16(Whh + (size_t)j0 * NH, /*lds_off=*/0u,
                  /*tile*/ 512u, 32u, 4u,
                  /*tensor*/ 512u, 512u, 4u,
                  /*d0_stride*/ 512ull, /*d1_stride*/ 512ull * 512ull);
    __builtin_amdgcn_s_wait_tensorcnt(0);
  }
#else
  for (int i = tid; i < (128 * 512) / 8; i += 256) {
    int r = i >> 6;
    int s = i & 63;
    int src_row = (r >> 5) * NH + j0 + (r & 31);
    *(v8bf*)&w_s[r * 512 + s * 8] = *(const v8bf*)&Whh[(size_t)src_row * NH + s * 8];
  }
#endif
  for (int i = tid; i < 32 * 32; i += 256) c_s[i] = 0.f;
  for (int i = tid; i < 32 * 32; i += 256) {
    int b = i >> 5, u = i & 31;
    h_ping[b * NH + j0 + u] = (bf16_t)0.f;
  }
  grid_sync(bar, goal);

  const int n0l = w * 16;  // this wave's 16 local gate columns

  for (int t = 0; t < NT; ++t) {
    const bf16_t* hsrc = (t & 1) ? h_pong : h_ping;
    bf16_t* hdst = (t & 1) ? h_ping : h_pong;

    // Stage full h_{t-1} [32 x 512] bf16 into LDS.
#if HAVE_TDM
    if (w == 0) {
      tdm_load_bf16(hsrc, /*lds_off=*/131072u,
                    /*tile*/ 512u, 32u, 0u,
                    /*tensor*/ 512u, 32u, 0u,
                    /*d0_stride*/ 512ull, /*d1_stride*/ 0ull);
      __builtin_amdgcn_s_wait_tensorcnt(0);
    }
#else
    for (int i = tid; i < (32 * 512) / 8; i += 256)
      *(v8bf*)&h_s[i * 8] = *(const v8bf*)&hsrc[i * 8];
#endif
    __syncthreads();

    // gates[32 x 128] = h @ Whh_slice^T : 2 M-tiles x (this wave's N-tile)
    v8f acc0 = {}, acc1 = {};
    const int kb = lh * 8;
    const int kb2 = lh * 8;
#pragma unroll 4
    for (int kk = 0; kk < 16; ++kk) {
      const bf16_t* p = &h_s[l15 * 512 + kk * 32 + kb];
      v16bf a0 = CAT16(*(const v8bf*)p, *(const v8bf*)(p + 16));
      const bf16_t* q = &h_s[(16 + l15) * 512 + kk * 32 + kb];
      v16bf a1 = CAT16(*(const v8bf*)q, *(const v8bf*)(q + 16));
      const bf16_t* bp = &w_s[(n0l + l15) * 512 + kk * 32 + kb2];
      v16bf bb = CAT16(*(const v8bf*)bp, *(const v8bf*)(bp + 16));
      acc0 = wmma_bf16(a0, bb, acc0);
      acc1 = wmma_bf16(a1, bb, acc1);
    }

    // Epilogue: add precomputed x-projection, park gates in LDS.
    const float* xpt = XP + (size_t)t * NB * NG;
    const int nl = n0l + l15;
    const int gcol = (nl >> 5) * NH + j0 + (nl & 31);  // global gate column
#pragma unroll
    for (int r = 0; r < 8; ++r) {
      const int M0 = r + 8 * lh;
      g_s[M0 * 128 + nl] = acc0[r] + xpt[(size_t)M0 * NG + gcol];
      g_s[(M0 + 16) * 128 + nl] = acc1[r] + xpt[(size_t)(M0 + 16) * NG + gcol];
    }
    __syncthreads();

    // Cell update for this WG's 32 units x 32 batches.
    for (int i = tid; i < 32 * 32; i += 256) {
      int b = i >> 5, u = i & 31;
      float gi = g_s[b * 128 + u];
      float gf = g_s[b * 128 + 32 + u];
      float gg = g_s[b * 128 + 64 + u];
      float go = g_s[b * 128 + 96 + u];
      float c = c_s[i];
      float nc = fsig(gf) * c + fsig(gi) * ftanh_(gg);
      float hv = fsig(go) * ftanh_(nc);
      c_s[i] = nc;
      bf16_t hb = (bf16_t)hv;
      hdst[b * NH + j0 + u] = hb;
      hs[((size_t)t * NB + b) * NH + j0 + u] = hb;
    }
    if (t + 1 < NT)
      __builtin_prefetch(XP + (size_t)(t + 1) * NB * NG + (wg * 128) + tid, 0, 1);
    grid_sync(bar, goal);
  }
}

// ------------------------------ final linear -------------------------------
__global__ void k_final(const bf16_t* __restrict__ hs, const float* __restrict__ Wout,
                        const float* __restrict__ bout, float* __restrict__ out) {
  int idx = blockIdx.x * blockDim.x + threadIdx.x;
  if (idx >= NB * NO) return;
  int b = idx / NO, o = idx % NO;
  const bf16_t* hrow = hs + ((size_t)(NT - 1) * NB + b) * NH;
  const float* wrow = Wout + (size_t)o * NH;
  float s = bout[o];
  for (int k = 0; k < NH; ++k) s += (float)hrow[k] * wrow[k];
  out[b * NO + o] = s;
}

// ------------------------------- launcher ----------------------------------
extern "C" void kernel_launch(void* const* d_in, const int* in_sizes, int n_in,
                              void* d_out, int out_size, void* d_ws, size_t ws_size,
                              hipStream_t stream) {
  (void)in_sizes; (void)n_in; (void)out_size; (void)ws_size;
  const float* X = (const float*)d_in[0];
  const float* Wih[3] = {(const float*)d_in[1], (const float*)d_in[5],
                         (const float*)d_in[9]};
  const float* Whh[3] = {(const float*)d_in[2], (const float*)d_in[6],
                         (const float*)d_in[10]};
  const float* bih[3] = {(const float*)d_in[3], (const float*)d_in[7],
                         (const float*)d_in[11]};
  const float* bhh[3] = {(const float*)d_in[4], (const float*)d_in[8],
                         (const float*)d_in[12]};
  const float* Wout = (const float*)d_in[13];
  const float* bout = (const float*)d_in[14];

  char* ws = (char*)d_ws;
  size_t off = 0;
  auto alloc = [&](size_t bytes) -> void* {
    void* p = ws + off;
    off = (off + bytes + 255) & ~(size_t)255;
    return p;
  };
  unsigned* bar = (unsigned*)alloc(256);
  bf16_t* hping = (bf16_t*)alloc((size_t)NB * NH * 2);
  bf16_t* hpong = (bf16_t*)alloc((size_t)NB * NH * 2);
  bf16_t* xin = (bf16_t*)alloc((size_t)BT * NI * 2);
  bf16_t* buf1 = (bf16_t*)alloc((size_t)BT * NH * 2);
  bf16_t* buf2 = (bf16_t*)alloc((size_t)BT * NH * 2);
  float* XP = (float*)alloc((size_t)BT * NG * 4);
  bf16_t* wih_b = (bf16_t*)alloc((size_t)3 * NG * NI * 2);
  bf16_t* whh_b = (bf16_t*)alloc((size_t)3 * NG * NH * 2);
  float* biasc = (float*)alloc((size_t)3 * NG * 4);

  k_init<<<1, 64, 0, stream>>>(bar);
  k_reshape_x<<<(NB * NC * NT * NF) / 256, 256, 0, stream>>>(X, xin);
  for (int l = 0; l < 3; ++l) {
    k_f32_to_bf16<<<(NG * NI) / 256, 256, 0, stream>>>(
        Wih[l], wih_b + (size_t)l * NG * NI, NG * NI);
    k_f32_to_bf16<<<(NG * NH) / 256, 256, 0, stream>>>(
        Whh[l], whh_b + (size_t)l * NG * NH, NG * NH);
    k_bias<<<NG / 256, 256, 0, stream>>>(bih[l], bhh[l], biasc + l * NG);
  }

  bf16_t* lin[3] = {xin, buf1, buf2};
  bf16_t* lout[3] = {buf1, buf2, buf1};
  dim3 ggrid(BT / 64, NG / 128);
  const size_t lds_bytes = 184320;  // 128K W + 32K h + 16K gates + 4K c
  for (int l = 0; l < 3; ++l) {
    k_xproj<<<ggrid, 256, 0, stream>>>(lin[l], wih_b + (size_t)l * NG * NI,
                                       biasc + l * NG, XP);
    k_lstm_seq<<<16, 256, lds_bytes, stream>>>(
        XP, whh_b + (size_t)l * NG * NH, hping, hpong, lout[l], bar,
        (unsigned)(l * (NT + 1) * 16));
  }
  k_final<<<(NB * NO + 255) / 256, 256, 0, stream>>>(buf1, Wout, bout,
                                                     (float*)d_out);
}
